// SimpleQuantizer_63900523430456
// MI455X (gfx1250) — matrix operand order repaired and checked by reading, not compiled, not run on hardware
//
#include <hip/hip_runtime.h>
#include <stdint.h>

typedef __bf16 v16bf __attribute__((ext_vector_type(16)));
typedef float  v8f   __attribute__((ext_vector_type(8)));

#define Bb 16
#define Dd 128
#define Tt 4096
#define Cc 16
#define Kk 1024
#define TM 128      // t-tile rows per workgroup
#define XRS 129     // padded fp32 LDS stride for residual
#define CH 128      // codes per LDS chunk
#define CQ 17       // uint4 stride per code row in LDS (136 bf16, padded)
#define NCHUNK (Cc * (Kk / CH))   // 128 staged chunks total

__device__ __forceinline__ unsigned short f2bf(float f) {
  uint32_t u = __builtin_bit_cast(uint32_t, f);
  u += 0x7FFFu + ((u >> 16) & 1u);          // round-to-nearest-even
  return (unsigned short)(u >> 16);
}

__device__ __forceinline__ uint32_t lds_off(const void* p) {
  // generic LDS pointer: low 32 bits hold the in-allocation LDS byte offset
  return (uint32_t)(uintptr_t)p;
}

__device__ __forceinline__ void async_b128(uint32_t lds_addr, uint32_t voff, uint64_t gbase) {
  asm volatile("global_load_async_to_lds_b128 %0, %1, %2"
               :: "v"(lds_addr), "v"(voff), "s"(gbase) : "memory");
}
__device__ __forceinline__ void async_b32(uint32_t lds_addr, uint32_t voff, uint64_t gbase) {
  asm volatile("global_load_async_to_lds_b32 %0, %1, %2"
               :: "v"(lds_addr), "v"(voff), "s"(gbase) : "memory");
}
__device__ __forceinline__ void wait_async0() {
  asm volatile("s_wait_asynccnt 0x0" ::: "memory");
}

// One-time prep: embed fp32 -> bf16 (for WMMA B operands) + per-code ||e||^2.
__global__ void __launch_bounds__(128)
rvq_prep(const float* __restrict__ embed, unsigned short* __restrict__ ebf,
         float* __restrict__ e2) {
  const int row = blockIdx.x;               // 0 .. C*K-1
  const int d   = threadIdx.x;              // 0 .. 127
  float f = embed[(size_t)row * Dd + d];
  ebf[(size_t)row * Dd + d] = f2bf(f);
  __shared__ float red[128];
  red[d] = f * f;
  __syncthreads();
  for (int s = 64; s > 0; s >>= 1) {
    if (d < s) red[d] += red[d + s];
    __syncthreads();
  }
  if (d == 0) e2[row] = red[0];
}

// Fused 16-stage residual VQ. One WG = 128 t-rows of one batch, residual in LDS.
// Wave w owns 32 rows (two 16-row A strips) and half the code N-tiles.
__global__ void __launch_bounds__(256)
rvq_main(const float* __restrict__ x, const float* __restrict__ embed,
         const unsigned short* __restrict__ ebf, const float* __restrict__ e2g,
         float* __restrict__ out_index, float* __restrict__ out_y) {
  __shared__ float xr[Dd * XRS];            // residual tile [d][t], fp32
  __shared__ uint4 codes[2][CH * CQ];       // double-buffered bf16 code chunks
  __shared__ float e2c[2][CH];
  __shared__ float redv[TM * 16];
  __shared__ int   redi[TM * 16];
  __shared__ int   idxrow[TM];

  const int tid  = threadIdx.x;
  const int lane = tid & 31;
  const int wave = tid >> 5;
  const int b    = blockIdx.x >> 5;         // 32 t-tiles per batch
  const int t0   = (blockIdx.x & 31) * TM;

  // Stage chunk gi (128 codes, bf16) + its ||e||^2 into buffer par via async DMA.
  auto stage = [&](int gi, int par) {
    const uint64_t gbase = (uint64_t)(uintptr_t)(ebf + ((size_t)gi * CH) * Dd);
    const uint32_t cbase = lds_off(&codes[par][0]);
    for (int i = tid; i < CH * 16; i += 256) {           // 2048 uint4, 8 per thread
      int row = i >> 4, seg = i & 15;
      async_b128(cbase + (uint32_t)(row * CQ + seg) * 16u, (uint32_t)i * 16u, gbase);
    }
    if (tid < CH) {
      const uint64_t g2 = (uint64_t)(uintptr_t)(e2g + (size_t)gi * CH);
      async_b32(lds_off(&e2c[par][0]) + (uint32_t)tid * 4u, (uint32_t)tid * 4u, g2);
    }
  };

  // Load x tile -> residual (coalesced along t)
  for (int i = tid; i < Dd * TM; i += 256) {
    int d = i >> 7, tl = i & (TM - 1);
    xr[d * XRS + tl] = x[((size_t)(b * Dd + d)) * Tt + t0 + tl];
  }
  stage(0, 0);                               // prologue: prefetch first chunk
  __syncthreads();

  const int rb    = (wave >> 1) * 32;        // this wave's 32-row block
  const int nhalf = wave & 1;                // which half of the N tiles
  const int mrow  = lane & 15;
  const int hi    = (lane >> 4) << 3;        // C/D layout: M = v + 8*(lane/16)
  const int kofs  = (lane < 16) ? 0 : 8;     // 16-bit A-matrix lane layout

  int gi = 0;                                // global chunk counter (0..127)
  for (int c = 0; c < Cc; ++c) {
    // Build bf16 A fragments for both strips (16x32 per k-step, K=128)
    union { v16bf v; unsigned short u[16]; } afr[2][4];
#pragma unroll
    for (int st = 0; st < 2; ++st) {
      const int myt = rb + st * 16 + mrow;
#pragma unroll
      for (int ks = 0; ks < 4; ++ks) {
#pragma unroll
        for (int j = 0; j < 8; ++j) {
          afr[st][ks].u[j]     = f2bf(xr[(ks * 32 + kofs + j)      * XRS + myt]);
          afr[st][ks].u[8 + j] = f2bf(xr[(ks * 32 + 16 + kofs + j) * XRS + myt]);
        }
      }
    }

    float minv[16]; int mini[16];            // [0..7]=strip0, [8..15]=strip1
#pragma unroll
    for (int v = 0; v < 16; ++v) { minv[v] = 3.4e38f; mini[v] = 0; }

    for (int ch = 0; ch < Kk / CH; ++ch, ++gi) {
      const int par = gi & 1;
      wait_async0();                         // chunk gi staged (issued last iter)
      __syncthreads();                       // visible to all waves; prev buffers free
      if (gi + 1 < NCHUNK) stage(gi + 1, par ^ 1);

      // 4 N-tiles for this wave; each B fragment feeds both strips (2 chains)
      for (int ntl = 0; ntl < 4; ++ntl) {
        const int nt = nhalf * 4 + ntl;
        v8f acc0 = {0.f, 0.f, 0.f, 0.f, 0.f, 0.f, 0.f, 0.f};
        v8f acc1 = {0.f, 0.f, 0.f, 0.f, 0.f, 0.f, 0.f, 0.f};
        const uint4* brow = &codes[par][(nt * 16 + mrow) * CQ];
#pragma unroll
        for (int ks = 0; ks < 4; ++ks) {
          union { v16bf v; uint4 q[2]; } bf;  // 16-bit B-matrix 32x16 layout
          int dimq = (ks * 32 + ((lane < 16) ? 0 : 8)) >> 3;  // uint4 = 8 bf16
          bf.q[0] = brow[dimq];  bf.q[1] = brow[dimq + 2];
          acc0 = __builtin_amdgcn_wmma_f32_16x16x32_bf16(
              false, afr[0][ks].v, false, bf.v, (short)0, acc0, false, false);
          acc1 = __builtin_amdgcn_wmma_f32_16x16x32_bf16(
              false, afr[1][ks].v, false, bf.v, (short)0, acc1, false, false);
        }
        const int   n   = (gi & 7) * CH + nt * 16 + mrow;  // code idx in codebook
        const float e2v = e2c[par][nt * 16 + mrow];
#pragma unroll
        for (int v = 0; v < 8; ++v) {        // score = ||e||^2 - 2 x.e
          float s0 = e2v - 2.0f * acc0[v];
          float s1 = e2v - 2.0f * acc1[v];
          if (s0 < minv[v])     { minv[v]     = s0; mini[v]     = n; }
          if (s1 < minv[8 + v]) { minv[8 + v] = s1; mini[8 + v] = n; }
        }
      }
    }

    // Argmin merge: even waves write, odd waves compare-merge (same rows, other N half)
    __syncthreads();
    if ((wave & 1) == 0) {
#pragma unroll
      for (int s = 0; s < 16; ++s) {
        int row = rb + (s >> 3) * 16 + (s & 7) + hi;
        redv[row * 16 + mrow] = minv[s];
        redi[row * 16 + mrow] = mini[s];
      }
    }
    __syncthreads();
    if (wave & 1) {
#pragma unroll
      for (int s = 0; s < 16; ++s) {
        int row = rb + (s >> 3) * 16 + (s & 7) + hi;
        float ov = redv[row * 16 + mrow];
        int   oi = redi[row * 16 + mrow];
        if (minv[s] < ov || (minv[s] == ov && mini[s] < oi)) {
          redv[row * 16 + mrow] = minv[s];
          redi[row * 16 + mrow] = mini[s];
        }
      }
    }
    __syncthreads();
    if (tid < TM) {
      float bv = redv[tid * 16]; int bi = redi[tid * 16];
#pragma unroll
      for (int s = 1; s < 16; ++s) {
        float val = redv[tid * 16 + s]; int ii = redi[tid * 16 + s];
        if (val < bv || (val == bv && ii < bi)) { bv = val; bi = ii; }
      }
      idxrow[tid] = bi;
      out_index[((size_t)(b * Cc + c)) * Tt + t0 + tid] = (float)bi;
    }
    __syncthreads();

    // Residual update in full fp32 from the original codebook (no bf16 drift)
    {
      int row = tid >> 1, half = tid & 1;
      const float* er = embed + ((size_t)(c * Kk + idxrow[row])) * Dd + half * 64;
#pragma unroll 8
      for (int j = 0; j < 64; ++j)
        xr[(half * 64 + j) * XRS + row] -= er[j];
    }
    __syncthreads();
  }

  // y = x - final residual
  for (int i = tid; i < Dd * TM; i += 256) {
    int d = i >> 7, tl = i & (TM - 1);
    size_t g = ((size_t)(b * Dd + d)) * Tt + t0 + tl;
    out_y[g] = x[g] - xr[d * XRS + tl];
  }
}

extern "C" void kernel_launch(void* const* d_in, const int* in_sizes, int n_in,
                              void* d_out, int out_size, void* d_ws, size_t ws_size,
                              hipStream_t stream) {
  const float* x     = (const float*)d_in[0];   // [B, D, T]
  const float* embed = (const float*)d_in[1];   // [C, K, D]
  unsigned short* ebf = (unsigned short*)d_ws;                       // 4 MB bf16 codes
  float* e2 = (float*)((char*)d_ws + (size_t)Cc * Kk * Dd * 2);      // 64 KB ||e||^2
  float* out_index = (float*)d_out;                                  // [B, C, T]
  float* out_y     = out_index + (size_t)Bb * Cc * Tt;               // [B, D, T]

  rvq_prep<<<Cc * Kk, 128, 0, stream>>>(embed, ebf, e2);
  rvq_main<<<Bb * (Tt / TM), 256, 0, stream>>>(x, embed, ebf, e2, out_index, out_y);
}
